// AdaptiveDensityTokenizer_49211735277869
// MI455X (gfx1250) — matrix operand order repaired and checked by reading, not compiled, not run on hardware
//
#include <hip/hip_runtime.h>
#include <hip/hip_bf16.h>
#include <math.h>

// Problem constants (match reference)
#define B_   2
#define K_   32768
#define C_   256
#define CH_  128      // C/2
#define D_   256
#define T_   256
#define RPD_ 3
#define R3_  27
#define KT_  (K_/16)  // M-tiles per batch for importance GEMM

typedef __attribute__((ext_vector_type(16))) _Float16 v16h;
typedef __attribute__((ext_vector_type(8)))  float    v8f;

// A-fragment K offset within a 32-wide K chunk (16-bit A, 16x32, ISA 7.12.2):
// lane<16 : halves 0..7 -> K 0..7,  halves 8..15 -> K 16..23
// lane>=16: halves 0..7 -> K 8..15, halves 8..15 -> K 24..31
__device__ __forceinline__ int a_kofs(int lane, int h) {
  int base = (h < 8) ? h : (h + 8);
  return (lane >= 16) ? (base + 8) : base;
}

// ---------------------------------------------------------------------------
// 0) zero the per-call accumulators (harness does not re-poison between runs)
// ---------------------------------------------------------------------------
__global__ void adt_init_kernel(int* counts, float* regImp, int* outIdx, int* filled) {
  int tid = threadIdx.x;
  for (int i = tid; i < B_ * R3_; i += 256) { counts[i] = 0; regImp[i] = 0.0f; }
  for (int i = tid; i < B_ * T_;  i += 256) { outIdx[i] = 0; filled[i] = 0; }
}

// ---------------------------------------------------------------------------
// 1) importance MLP: imp = softplus(relu(X@W1+b1)@W2+b2)   [WMMA f16->f32]
//    one wave per 16 rows of K; hidden (16x128) lives entirely in 8 v8f accs
// ---------------------------------------------------------------------------
__global__ __launch_bounds__(32) void adt_importance_kernel(
    const float* __restrict__ features, const float* __restrict__ W1,
    const float* __restrict__ b1, const float* __restrict__ W2,
    const float* __restrict__ b2, float* __restrict__ imp) {
  int blk  = blockIdx.x;
  int b    = blk / KT_;
  int row0 = (blk % KT_) * 16;
  int lane = threadIdx.x;
  int mlo  = lane & 15;
  int p    = row0 + mlo;                       // this lane's A-matrix row (point id)
  const float* fb = features + (size_t)b * C_ * K_;

  v8f zero = {};
  v8f acc[8];
#pragma unroll
  for (int i = 0; i < 8; ++i) acc[i] = zero;

  for (int kk = 0; kk < 8; ++kk) {             // K(=C) in chunks of 32
    v16h a;
#pragma unroll
    for (int h = 0; h < 16; ++h) {
      int cA = kk * 32 + a_kofs(lane, h);
      a[h] = (_Float16)fb[(size_t)cA * K_ + p];  // feat_bkc[b,p,cA] (strided gather)
    }

#pragma unroll
    for (int nt = 0; nt < 8; ++nt) {           // 8 N-tiles cover hidden dim 128
      v16h w;
#pragma unroll
      for (int h = 0; h < 16; ++h)
        w[h] = (_Float16)W1[(kk * 32 + a_kofs(lane, h)) * CH_ + nt * 16 + mlo];
      acc[nt] = __builtin_amdgcn_wmma_f32_16x16x32_f16(
          false, a, false, w, (short)0, acc[nt], false, false);
    }
  }

  // fuse bias + relu + (h @ W2): lane holds column n = nt*16+mlo for 8 M rows
  float part[8];
#pragma unroll
  for (int r = 0; r < 8; ++r) part[r] = 0.0f;
#pragma unroll
  for (int nt = 0; nt < 8; ++nt) {
    int n = nt * 16 + mlo;
    float bb = b1[n], w2 = W2[n];
#pragma unroll
    for (int r = 0; r < 8; ++r) {
      float hv = acc[nt][r] + bb;
      hv = fmaxf(hv, 0.0f);
      part[r] += hv * w2;
    }
  }
  // reduce across the 16 lanes that share the same M rows
#pragma unroll
  for (int off = 1; off < 16; off <<= 1) {
#pragma unroll
    for (int r = 0; r < 8; ++r) part[r] += __shfl_xor(part[r], off, 16);
  }
  if (mlo == 0) {
    int mhi = (lane >= 16) ? 8 : 0;
    float bias2 = b2[0];
#pragma unroll
    for (int r = 0; r < 8; ++r) {
      float x = part[r] + bias2;
      float sp = (x > 20.0f) ? x : log1pf(expf(x));   // softplus
      imp[(size_t)b * K_ + row0 + mhi + r] = sp;
    }
  }
}

// ---------------------------------------------------------------------------
// 2) per-batch xyz min/max (one 256-thread block per batch)
// ---------------------------------------------------------------------------
__global__ void adt_minmax_kernel(const float* __restrict__ xyz, float* __restrict__ minmax) {
  __shared__ float s[6][256];
  int b = blockIdx.x, tid = threadIdx.x;
  const float* xb = xyz + (size_t)b * K_ * 3;
  float mn0 = 3e38f, mn1 = 3e38f, mn2 = 3e38f;
  float mx0 = -3e38f, mx1 = -3e38f, mx2 = -3e38f;
  for (int k = tid; k < K_; k += 256) {
    float x = xb[k * 3 + 0], y = xb[k * 3 + 1], z = xb[k * 3 + 2];
    mn0 = fminf(mn0, x); mn1 = fminf(mn1, y); mn2 = fminf(mn2, z);
    mx0 = fmaxf(mx0, x); mx1 = fmaxf(mx1, y); mx2 = fmaxf(mx2, z);
  }
  s[0][tid] = mn0; s[1][tid] = mn1; s[2][tid] = mn2;
  s[3][tid] = mx0; s[4][tid] = mx1; s[5][tid] = mx2;
  __syncthreads();
  for (int st = 128; st > 0; st >>= 1) {
    if (tid < st) {
      s[0][tid] = fminf(s[0][tid], s[0][tid + st]);
      s[1][tid] = fminf(s[1][tid], s[1][tid + st]);
      s[2][tid] = fminf(s[2][tid], s[2][tid + st]);
      s[3][tid] = fmaxf(s[3][tid], s[3][tid + st]);
      s[4][tid] = fmaxf(s[4][tid], s[4][tid + st]);
      s[5][tid] = fmaxf(s[5][tid], s[5][tid + st]);
    }
    __syncthreads();
  }
  if (tid == 0) {
    minmax[b * 6 + 0] = s[0][0]; minmax[b * 6 + 1] = s[1][0]; minmax[b * 6 + 2] = s[2][0];
    minmax[b * 6 + 3] = s[3][0]; minmax[b * 6 + 4] = s[4][0]; minmax[b * 6 + 5] = s[5][0];
  }
}

// ---------------------------------------------------------------------------
// 3) region id per point + counts/region-importance histograms
// ---------------------------------------------------------------------------
__global__ void adt_region_kernel(const float* __restrict__ xyz,
                                  const float* __restrict__ minmax,
                                  const float* __restrict__ imp,
                                  int* __restrict__ rid,
                                  int* __restrict__ counts,
                                  float* __restrict__ regImp) {
  int idx = blockIdx.x * blockDim.x + threadIdx.x;
  if (idx >= B_ * K_) return;
  int b = idx / K_;
  const float* p = xyz + (size_t)idx * 3;
  float v[3] = { p[0], p[1], p[2] };
  bool valid = (fabsf(v[0]) + fabsf(v[1]) + fabsf(v[2])) > 0.0f;
  int ri[3];
#pragma unroll
  for (int d = 0; d < 3; ++d) {
    float mn = minmax[b * 6 + d], mx = minmax[b * 6 + 3 + d];
    float nrm = (v[d] - mn) / (mx - mn + 1e-6f);
    float s = fminf(fmaxf(nrm * (float)RPD_, 0.0f), (float)(RPD_ - 1));
    ri[d] = (int)s;
  }
  int r = ri[0] * 9 + ri[1] * 3 + ri[2];
  rid[idx] = valid ? r : R3_;
  if (valid) {
    atomicAdd(&counts[b * R3_ + r], 1);
    atomicAdd(&regImp[b * R3_ + r], imp[idx]);
  }
}

// ---------------------------------------------------------------------------
// 4) token budget per region: n_r, c_r, exclusive start offsets
// ---------------------------------------------------------------------------
__global__ void adt_plan_kernel(const int* __restrict__ counts,
                                const float* __restrict__ regImp,
                                int* __restrict__ nr, int* __restrict__ cr,
                                int* __restrict__ startv) {
  int b = blockIdx.x;
  if (threadIdx.x != 0) return;
  float s = 0.0f;
  for (int r = 0; r < R3_; ++r) s += regImp[b * R3_ + r];
  int acc = 0;
  for (int r = 0; r < R3_; ++r) {
    int n = (int)rintf(regImp[b * R3_ + r] / (s + 1e-8f) * (float)T_);  // round-half-even
    int cnt = counts[b * R3_ + r];
    int c = (n == 0) ? 0 : ((n < cnt) ? n : cnt);
    nr[b * R3_ + r] = n;
    cr[b * R3_ + r] = c;
    startv[b * R3_ + r] = acc;
    acc += c;
  }
}

// ---------------------------------------------------------------------------
// 5) ascending-index list per region (stable: index order) — first T_ entries
// ---------------------------------------------------------------------------
__global__ __launch_bounds__(32) void adt_asc_kernel(const int* __restrict__ rid,
                                                     int* __restrict__ ascb) {
  __shared__ int flags[32];
  int br = blockIdx.x;
  int b = br / R3_, r = br % R3_;
  int lane = threadIdx.x;
  const int* rb = rid + (size_t)b * K_;
  int base = 0;
  for (int k0 = 0; k0 < K_; k0 += 32) {
    if (base >= T_) break;
    int k = k0 + lane;
    int f = (rb[k] == r) ? 1 : 0;
    flags[lane] = f;
    __syncthreads();
    int pre = 0, tot = 0;
    for (int i = 0; i < 32; ++i) { if (i < lane) pre += flags[i]; tot += flags[i]; }
    if (f) {
      int pos = base + pre;
      if (pos < T_) ascb[br * T_ + pos] = k;
    }
    base += tot;
    __syncthreads();
  }
}

// ---------------------------------------------------------------------------
// 6) init per-(b,r) min-distance arrays for FPS
// ---------------------------------------------------------------------------
__global__ void adt_mind_init_kernel(const int* __restrict__ rid, float* __restrict__ mind) {
  int idx = blockIdx.x * blockDim.x + threadIdx.x;
  if (idx >= B_ * R3_ * K_) return;
  int br = idx / K_, k = idx % K_;
  int b = br / R3_, r = br % R3_;
  mind[idx] = (rid[(size_t)b * K_ + k] == r) ? 1e10f : -1.0f;
}

// ---------------------------------------------------------------------------
// 7) masked farthest-point sampling: one block per (batch, region)
//    argmax tie-break = lowest index (matches jnp.argmax)
// ---------------------------------------------------------------------------
__global__ void adt_fps_kernel(const float* __restrict__ xyz,
                               float* __restrict__ mind, int* __restrict__ fpsIdx) {
  __shared__ float sv[256];
  __shared__ int   si[256];
  __shared__ float ppt[3];
  int br = blockIdx.x;
  int b = br / R3_;
  float* md = mind + (size_t)br * K_;
  const float* xb = xyz + (size_t)b * K_ * 3;
  int tid = threadIdx.x;

  for (int step = 0; step < T_; ++step) {
    float bv = -3e38f; int bi = 0;
    for (int k = tid; k < K_; k += 256) {
      float v = md[k];
      if (v > bv) { bv = v; bi = k; }          // strict > keeps lowest index per thread
    }
    sv[tid] = bv; si[tid] = bi;
    __syncthreads();
    for (int st = 128; st > 0; st >>= 1) {
      if (tid < st) {
        float v2 = sv[tid + st]; int i2 = si[tid + st];
        if (v2 > sv[tid] || (v2 == sv[tid] && i2 < si[tid])) { sv[tid] = v2; si[tid] = i2; }
      }
      __syncthreads();
    }
    int pick = si[0];
    if (tid == 0) {
      fpsIdx[br * T_ + step] = pick;
      ppt[0] = xb[pick * 3 + 0]; ppt[1] = xb[pick * 3 + 1]; ppt[2] = xb[pick * 3 + 2];
    }
    __syncthreads();
    float px = ppt[0], py = ppt[1], pz = ppt[2];
    for (int k = tid; k < K_; k += 256) {
      float dx = xb[k * 3 + 0] - px, dy = xb[k * 3 + 1] - py, dz = xb[k * 3 + 2] - pz;
      float d = dx * dx + dy * dy + dz * dz;
      float v = md[k];
      md[k] = (d < v) ? d : v;
    }
    __syncthreads();
  }
}

// ---------------------------------------------------------------------------
// 8) select asc vs fps per region and scatter into the packed token list
// ---------------------------------------------------------------------------
__global__ void adt_select_kernel(const int* __restrict__ counts, const int* __restrict__ nr,
                                  const int* __restrict__ cr, const int* __restrict__ startv,
                                  const int* __restrict__ ascb, const int* __restrict__ fpsIdx,
                                  int* __restrict__ outIdx, int* __restrict__ filled) {
  int br = blockIdx.x;
  int b = br / R3_, r = br % R3_;
  int j = threadIdx.x;
  int c = cr[b * R3_ + r];
  if (j < c) {
    int pos = startv[b * R3_ + r] + j;
    if (pos < T_) {
      int useAsc = (counts[b * R3_ + r] <= nr[b * R3_ + r]);
      int s = useAsc ? ascb[br * T_ + j] : fpsIdx[br * T_ + j];
      outIdx[b * T_ + pos] = s;
      filled[b * T_ + pos] = 1;
    }
  }
}

// ---------------------------------------------------------------------------
// 9) xyz_tok output (masked gather)
// ---------------------------------------------------------------------------
__global__ void adt_xyz_out_kernel(const float* __restrict__ xyz, const int* __restrict__ outIdx,
                                   const int* __restrict__ filled, float* __restrict__ out) {
  int idx = blockIdx.x * blockDim.x + threadIdx.x;
  if (idx >= B_ * T_) return;
  int b = idx / T_;
  int f = filled[idx];
  int k = outIdx[idx];
#pragma unroll
  for (int d = 0; d < 3; ++d)
    out[idx * 3 + d] = f ? xyz[((size_t)b * K_ + k) * 3 + d] : 0.0f;
}

// ---------------------------------------------------------------------------
// 10) token projection: feat_tok[b,d,t] = mask * (gather(feat) @ Wa + ba)
//     one wave per 16x16 output tile, WMMA f16->f32
// ---------------------------------------------------------------------------
__global__ __launch_bounds__(32) void adt_token_gemm_kernel(
    const float* __restrict__ features, const float* __restrict__ Wa,
    const float* __restrict__ ba, const int* __restrict__ outIdx,
    const int* __restrict__ filled, float* __restrict__ out) {
  int blk = blockIdx.x;
  int b = blk / 256;
  int tile = blk % 256;
  int t0 = (tile >> 4) * 16;   // M tile (tokens)
  int n0 = (tile & 15) * 16;   // N tile (token dim)
  int lane = threadIdx.x;
  int mlo = lane & 15;
  int point = outIdx[b * T_ + t0 + mlo];       // A-row gather index for this lane
  const float* fb = features + (size_t)b * C_ * K_;

  v8f acc = {};
  for (int kk = 0; kk < 8; ++kk) {             // K(=C) chunks of 32
    v16h a, w;
#pragma unroll
    for (int h = 0; h < 16; ++h) {
      int cA = kk * 32 + a_kofs(lane, h);
      a[h] = (_Float16)fb[(size_t)cA * K_ + point];
      int cB = kk * 32 + a_kofs(lane, h);
      w[h] = (_Float16)Wa[cB * D_ + n0 + mlo];
    }
    acc = __builtin_amdgcn_wmma_f32_16x16x32_f16(
        false, a, false, w, (short)0, acc, false, false);
  }

  int n = n0 + mlo;
  float bav = ba[n];
  int mhi = (lane >= 16) ? 8 : 0;
  const size_t xyzWords = (size_t)B_ * T_ * 3;           // 1536
#pragma unroll
  for (int r = 0; r < 8; ++r) {
    int tt = t0 + r + mhi;
    float v = filled[b * T_ + tt] ? (acc[r] + bav) : 0.0f;
    out[xyzWords + ((size_t)(b * D_ + n)) * T_ + tt] = v; // (B,D,T) layout
  }
}

// ---------------------------------------------------------------------------
extern "C" void kernel_launch(void* const* d_in, const int* in_sizes, int n_in,
                              void* d_out, int out_size, void* d_ws, size_t ws_size,
                              hipStream_t stream) {
  (void)in_sizes; (void)n_in; (void)out_size; (void)ws_size;

  const float* xyz      = (const float*)d_in[0];
  const float* features = (const float*)d_in[1];
  const float* W1 = (const float*)d_in[2];
  const float* b1 = (const float*)d_in[3];
  const float* W2 = (const float*)d_in[4];
  const float* b2 = (const float*)d_in[5];
  const float* Wa = (const float*)d_in[6];
  const float* ba = (const float*)d_in[7];
  float* out = (float*)d_out;

  // workspace carve-up (bytes)
  char* ws = (char*)d_ws;
  size_t o = 0;
  float* imp    = (float*)(ws + o); o += (size_t)B_ * K_ * 4;
  int*   rid    = (int*)  (ws + o); o += (size_t)B_ * K_ * 4;
  float* minmax = (float*)(ws + o); o += (size_t)B_ * 6 * 4;
  int*   counts = (int*)  (ws + o); o += (size_t)B_ * R3_ * 4;
  float* regImp = (float*)(ws + o); o += (size_t)B_ * R3_ * 4;
  int*   nr     = (int*)  (ws + o); o += (size_t)B_ * R3_ * 4;
  int*   cr     = (int*)  (ws + o); o += (size_t)B_ * R3_ * 4;
  int*   startv = (int*)  (ws + o); o += (size_t)B_ * R3_ * 4;
  int*   ascb   = (int*)  (ws + o); o += (size_t)B_ * R3_ * T_ * 4;
  int*   fpsIdx = (int*)  (ws + o); o += (size_t)B_ * R3_ * T_ * 4;
  int*   outIdx = (int*)  (ws + o); o += (size_t)B_ * T_ * 4;
  int*   filled = (int*)  (ws + o); o += (size_t)B_ * T_ * 4;
  float* mind   = (float*)(ws + o); o += (size_t)B_ * R3_ * K_ * 4;

  adt_init_kernel<<<1, 256, 0, stream>>>(counts, regImp, outIdx, filled);
  adt_importance_kernel<<<B_ * KT_, 32, 0, stream>>>(features, W1, b1, W2, b2, imp);
  adt_minmax_kernel<<<B_, 256, 0, stream>>>(xyz, minmax);
  adt_region_kernel<<<(B_ * K_ + 255) / 256, 256, 0, stream>>>(xyz, minmax, imp, rid, counts, regImp);
  adt_plan_kernel<<<B_, 32, 0, stream>>>(counts, regImp, nr, cr, startv);
  adt_asc_kernel<<<B_ * R3_, 32, 0, stream>>>(rid, ascb);
  adt_mind_init_kernel<<<(B_ * R3_ * K_ + 255) / 256, 256, 0, stream>>>(rid, mind);
  adt_fps_kernel<<<B_ * R3_, 256, 0, stream>>>(xyz, mind, fpsIdx);
  adt_select_kernel<<<B_ * R3_, 256, 0, stream>>>(counts, nr, cr, startv, ascb, fpsIdx, outIdx, filled);
  adt_xyz_out_kernel<<<(B_ * T_ + 255) / 256, 256, 0, stream>>>(xyz, outIdx, filled, out);
  adt_token_gemm_kernel<<<B_ * 256, 32, 0, stream>>>(features, Wa, ba, outIdx, filled, out);
}
